// New_GAU_16776142258403
// MI455X (gfx1250) — matrix operand order repaired and checked by reading, not compiled, not run on hardware
//
#include <hip/hip_runtime.h>
#include <hip/hip_bf16.h>
#include <math.h>

typedef __bf16 bf16_t;
typedef __attribute__((ext_vector_type(16))) __bf16 v16bf;
typedef __attribute__((ext_vector_type(8)))  __bf16 v8bf;
typedef __attribute__((ext_vector_type(8)))  float  v8f;

#define WMMA_BF16(a, b, c) \
  __builtin_amdgcn_wmma_f32_16x16x32_bf16(false, (a), false, (b), (short)0, (c), false, false)

// ---------- constants for this problem ----------
static constexpr int BB   = 4;
static constexpr int NN   = 16384;   // 128*128 tokens
static constexpr int ROWS = BB * NN; // 65536
static constexpr float INV_N = 1.0f / 16384.0f;

// ---------- fragment helpers (layouts per CDNA5 ISA 7.12.2) ----------
// build a 16-element bf16 fragment from two contiguous 8-element (16B) groups
__device__ inline v16bf frag_from2(const bf16_t* p0, const bf16_t* p1) {
  v8bf a = *(const v8bf*)p0;
  v8bf b = *(const v8bf*)p1;
  v16bf r;
#pragma unroll
  for (int i = 0; i < 8; ++i) { r[i] = a[i]; r[i + 8] = b[i]; }
  return r;
}
// A 16x32 bf16: lane row-major, e0..7 -> k = kbase+8*hi+e ; e8..15 -> k = kbase+16+8*hi+(e-8)
__device__ inline v16bf load_fragA(const bf16_t* row, int kbase, int hi) {
  return frag_from2(row + kbase + 8 * hi, row + kbase + 16 + 8 * hi);
}
// B 32x16 bf16 (stored N-major, K contiguous per lane row): e0..7 -> k = kbase+8*hi+e ; e8..15 -> k = kbase+16+8*hi+(e-8)
__device__ inline v16bf load_fragB(const bf16_t* row, int kbase, int hi) {
  return frag_from2(row + kbase + 8 * hi, row + kbase + 16 + 8 * hi);
}

// fast transcendental path: v_exp_f32 + v_rcp_f32 instead of libm expf + IEEE divide
__device__ inline float silu_f(float x) {
  return x * __builtin_amdgcn_rcpf(1.0f + __expf(-x));
}
__device__ inline float gelu_f(float x) { return 0.5f * x * (1.0f + erff(x * 0.70710678118654752f)); }

// ---------- weight prep ----------
__global__ void transpose_bf16(const float* src, bf16_t* dst, int R, int Cl) {
  int i = blockIdx.x * blockDim.x + threadIdx.x;
  if (i >= R * Cl) return;
  int r = i / Cl, c = i % Cl;
  dst[(size_t)c * R + r] = (bf16_t)src[i];
}

// src layout (O=96, C=96, s, s) -> dst[o][(u*s+v)*96 + c]
__global__ void conv_w_reorder(const float* src, bf16_t* dst, int ss) {
  int K = 96 * ss;
  int i = blockIdx.x * blockDim.x + threadIdx.x;
  if (i >= 96 * K) return;
  int o = i / K, rem = i % K;
  int uv = rem / 96, c = rem % 96;
  dst[i] = (bf16_t)src[(size_t)(o * 96 + c) * ss + uv];
}

// ---------- LayerNorm over C=96 (one wave per row), optional exact gelu ----------
__global__ void ln_kernel(const float* __restrict__ x, const float* __restrict__ w,
                          const float* __restrict__ b, bf16_t* __restrict__ out,
                          int nrows, int do_gelu) {
  int warps = blockDim.x >> 5;
  int row = blockIdx.x * warps + (threadIdx.x >> 5);
  if (row >= nrows) return;
  int lane = threadIdx.x & 31;
  const float* xr = x + (size_t)row * 96;
  float v0 = xr[lane], v1 = xr[lane + 32], v2 = xr[lane + 64];
  float s = v0 + v1 + v2;
#pragma unroll
  for (int o = 16; o; o >>= 1) s += __shfl_xor(s, o, 32);
  float m = s * (1.0f / 96.0f);
  v0 -= m; v1 -= m; v2 -= m;
  float q = v0 * v0 + v1 * v1 + v2 * v2;
#pragma unroll
  for (int o = 16; o; o >>= 1) q += __shfl_xor(q, o, 32);
  float ri = __builtin_amdgcn_rsqf(q * (1.0f / 96.0f) + 1e-5f);
  float y0 = v0 * ri * w[lane]      + b[lane];
  float y1 = v1 * ri * w[lane + 32] + b[lane + 32];
  float y2 = v2 * ri * w[lane + 64] + b[lane + 64];
  if (do_gelu) { y0 = gelu_f(y0); y1 = gelu_f(y1); y2 = gelu_f(y2); }
  bf16_t* orow = out + (size_t)row * 96;
  orow[lane] = (bf16_t)y0; orow[lane + 32] = (bf16_t)y1; orow[lane + 64] = (bf16_t)y2;
}

// ---------- h = silu(nx @ Wh + bh): split into v1/v2 (bf16, padded 64) + gate (f32) ----------
__global__ void gemm_h(const bf16_t* __restrict__ A, const bf16_t* __restrict__ Bt,
                       const float* __restrict__ bh, bf16_t* __restrict__ v1,
                       bf16_t* __restrict__ v2, float* __restrict__ gate) {
  int wave = blockIdx.x * (blockDim.x >> 5) + (threadIdx.x >> 5);
  int nt = wave % 12, mt = wave / 12;           // 192/16 = 12 col tiles
  if (mt >= ROWS / 16) return;
  int lane = threadIdx.x & 31, lm = lane & 15, hi = lane >> 4;
  const bf16_t* arow = A  + (size_t)(mt * 16 + lm) * 96;
  const bf16_t* brow = Bt + (size_t)(nt * 16 + lm) * 96;
  v8f acc = {};
#pragma unroll
  for (int kt = 0; kt < 3; ++kt)
    acc = WMMA_BF16(load_fragA(arow, kt * 32, hi), load_fragB(brow, kt * 32, hi), acc);
  int col = nt * 16 + lm;
  float bias = bh[col];
#pragma unroll
  for (int r = 0; r < 8; ++r) {
    size_t row = (size_t)mt * 16 + r + 8 * hi;
    float sv = silu_f(acc[r] + bias);
    if (col < 48)       v1[row * 64 + col]        = (bf16_t)sv;
    else if (col < 96)  v2[row * 64 + (col - 48)] = (bf16_t)sv;
    else                gate[row * 96 + (col - 96)] = sv;
  }
}

// ---------- conv patch-embed as im2col GEMM (K = 96*s*s), +bias, f32 out ----------
__global__ void conv_gemm(const bf16_t* __restrict__ nx, const bf16_t* __restrict__ wr,
                          const float* __restrict__ cb, float* __restrict__ out,
                          int s, int T, int Ho, int K) {
  int wave = blockIdx.x * (blockDim.x >> 5) + (threadIdx.x >> 5);
  int nt = wave % 6, mt = wave / 6;             // 96/16 = 6 col tiles
  int mtiles = (BB * T) >> 4;
  if (mt >= mtiles) return;
  int lane = threadIdx.x & 31, lm = lane & 15, hi = lane >> 4;
  int g = mt * 16 + lm;
  int b = g / T, t = g % T, oi = t / Ho, oj = t % Ho;
  const bf16_t* nxb = nx + (size_t)b * NN * 96;
  int px = s * oi * 128 + s * oj;               // base pixel (linear) of the patch
  const bf16_t* brow = wr + (size_t)(nt * 16 + lm) * K;
  v8f acc = {};
  for (int kt = 0; kt < K; kt += 32) {
    int k0 = kt + 8 * hi;
    int uv0 = k0 / 96, c0 = k0 % 96;
    const bf16_t* p0 = nxb + (size_t)(px + (uv0 / s) * 128 + (uv0 % s)) * 96 + c0;
    int k1 = k0 + 16;
    int uv1 = k1 / 96, c1 = k1 % 96;
    const bf16_t* p1 = nxb + (size_t)(px + (uv1 / s) * 128 + (uv1 % s)) * 96 + c1;
    acc = WMMA_BF16(frag_from2(p0, p1), load_fragB(brow, kt, hi), acc);
  }
  int col = nt * 16 + lm;
  float bias = cb[col];
#pragma unroll
  for (int r = 0; r < 8; ++r)
    out[(size_t)(mt * 16 + r + 8 * hi) * 96 + col] = acc[r] + bias;
}

// ---------- Z = silu(x @ WqkT + bqk); q = Z*g[0]+be[0] (token-major, pad 64);
//            k = Z*g[1]+be[1] stored TRANSPOSED (dim-major, stride T) ----------
__global__ void qk_kernel(const bf16_t* __restrict__ xb, const bf16_t* __restrict__ WqkT,
                          const float* __restrict__ bqk, const float* __restrict__ gv,
                          const float* __restrict__ bev, bf16_t* __restrict__ qpad,
                          bf16_t* __restrict__ kT, int T, int nrows) {
  int wave = blockIdx.x * (blockDim.x >> 5) + (threadIdx.x >> 5);
  int nt = wave % 3, mt = wave / 3;             // 48/16 = 3 col tiles
  if (mt * 16 >= nrows) return;
  int lane = threadIdx.x & 31, lm = lane & 15, hi = lane >> 4;
  const bf16_t* arow = xb   + (size_t)(mt * 16 + lm) * 96;
  const bf16_t* brow = WqkT + (size_t)(nt * 16 + lm) * 96;
  v8f acc = {};
#pragma unroll
  for (int kt = 0; kt < 3; ++kt)
    acc = WMMA_BF16(load_fragA(arow, kt * 32, hi), load_fragB(brow, kt * 32, hi), acc);
  int col = nt * 16 + lm;
  float bias = bqk[col];
  float gq = gv[col], bq = bev[col], gk = gv[48 + col], bk = bev[48 + col];
#pragma unroll
  for (int r = 0; r < 8; ++r) {
    int grow = mt * 16 + r + 8 * hi;
    float z = silu_f(acc[r] + bias);
    qpad[(size_t)grow * 64 + col] = (bf16_t)(z * gq + bq);
    int b = grow / T, t = grow % T;
    kT[((size_t)b * 48 + col) * T + t] = (bf16_t)(z * gk + bk);
  }
}

// ---------- fused linear attention:  V = relu(v·qᵀ/N)² · k  (per 16-row tile) ----------
__global__ void attn_kernel(const bf16_t* __restrict__ vpad, const bf16_t* __restrict__ qpad,
                            const bf16_t* __restrict__ kT, float* __restrict__ xc,
                            int M, int colofs) {
  __shared__ __align__(16) bf16_t lds[4][16 * 32];
  int wid = threadIdx.x >> 5;
  int wave = blockIdx.x * 4 + wid;
  int lane = threadIdx.x & 31, lm = lane & 15, hi = lane >> 4;
  size_t r0 = (size_t)wave * 16;                // global row base (grid sized exactly)
  int b = (int)(r0 >> 14);                      // r0 / 16384
  const bf16_t* arow = vpad + (r0 + lm) * 64;
  v16bf av0 = load_fragA(arow, 0, hi);          // v dims 0..31
  v16bf av1 = load_fragA(arow, 32, hi);         // v dims 32..63 (48..63 are zero pad)
  const bf16_t* qb = qpad + (size_t)b * M * 64;
  const bf16_t* kb = kT   + (size_t)b * 48 * M;
  bf16_t* myl = lds[wid];
  v8f vacc0 = {}, vacc1 = {}, vacc2 = {};
  for (int mc = 0; mc < M; mc += 32) {
#pragma unroll
    for (int st = 0; st < 2; ++st) {
      const bf16_t* qrow = qb + (size_t)(mc + st * 16 + lm) * 64;
      v8f aacc = {};
      aacc = WMMA_BF16(av0, load_fragB(qrow, 0, hi), aacc);
      aacc = WMMA_BF16(av1, load_fragB(qrow, 32, hi), aacc);
#pragma unroll
      for (int r = 0; r < 8; ++r) {             // relu(a/N)^2, relayout D -> row-major LDS
        float sv = aacc[r] * INV_N;
        sv = sv > 0.0f ? sv * sv : 0.0f;
        myl[(r + 8 * hi) * 32 + st * 16 + lm] = (bf16_t)sv;
      }
    }
    __syncthreads();
    v16bf af = load_fragA(myl + lm * 32, 0, hi);         // A1 fragment 16x32
    const bf16_t* k0p = kb + (size_t)(0  + lm) * M + mc;
    const bf16_t* k1p = kb + (size_t)(16 + lm) * M + mc;
    const bf16_t* k2p = kb + (size_t)(32 + lm) * M + mc;
    vacc0 = WMMA_BF16(af, load_fragB(k0p, 0, hi), vacc0);
    vacc1 = WMMA_BF16(af, load_fragB(k1p, 0, hi), vacc1);
    vacc2 = WMMA_BF16(af, load_fragB(k2p, 0, hi), vacc2);
    __syncthreads();
  }
#pragma unroll
  for (int r = 0; r < 8; ++r) {
    float* o = xc + (r0 + r + 8 * hi) * 96 + colofs;
    o[lm] = vacc0[r]; o[16 + lm] = vacc1[r]; o[32 + lm] = vacc2[r];
  }
}

// ---------- value = xc * gate (to bf16) ----------
__global__ void value_mul(const float* __restrict__ xc, const float* __restrict__ gate,
                          bf16_t* __restrict__ valb, int n) {
  int i = blockIdx.x * blockDim.x + threadIdx.x;
  if (i < n) valb[i] = (bf16_t)(xc[i] * gate[i]);
}

// ---------- out = value @ proj + proj_b + xc ----------
__global__ void proj_kernel(const bf16_t* __restrict__ A, const bf16_t* __restrict__ Bt,
                            const float* __restrict__ pb, const float* __restrict__ xc,
                            float* __restrict__ out) {
  int wave = blockIdx.x * (blockDim.x >> 5) + (threadIdx.x >> 5);
  int nt = wave % 6, mt = wave / 6;
  if (mt >= ROWS / 16) return;
  int lane = threadIdx.x & 31, lm = lane & 15, hi = lane >> 4;
  const bf16_t* arow = A  + (size_t)(mt * 16 + lm) * 96;
  const bf16_t* brow = Bt + (size_t)(nt * 16 + lm) * 96;
  v8f acc = {};
#pragma unroll
  for (int kt = 0; kt < 3; ++kt)
    acc = WMMA_BF16(load_fragA(arow, kt * 32, hi), load_fragB(brow, kt * 32, hi), acc);
  int col = nt * 16 + lm;
  float bias = pb[col];
#pragma unroll
  for (int r = 0; r < 8; ++r) {
    size_t row = (size_t)mt * 16 + r + 8 * hi;
    out[row * 96 + col] = acc[r] + bias + xc[row * 96 + col];
  }
}

// =======================================================================
extern "C" void kernel_launch(void* const* d_in, const int* in_sizes, int n_in,
                              void* d_out, int out_size, void* d_ws, size_t ws_size,
                              hipStream_t stream) {
  const float* x      = (const float*)d_in[0];
  const float* norm_w = (const float*)d_in[3];
  const float* norm_b = (const float*)d_in[4];
  const float* Wh     = (const float*)d_in[5];
  const float* bh     = (const float*)d_in[6];
  const float* Wqk    = (const float*)d_in[7];
  const float* bqk    = (const float*)d_in[8];
  const float* g1     = (const float*)d_in[9];
  const float* be1    = (const float*)d_in[10];
  const float* g2     = (const float*)d_in[11];
  const float* be2    = (const float*)d_in[12];
  const float* sr1_w  = (const float*)d_in[13];
  const float* sr1_b  = (const float*)d_in[14];
  const float* sr2_w  = (const float*)d_in[15];
  const float* sr2_b  = (const float*)d_in[16];
  const float* n1_w   = (const float*)d_in[17];
  const float* n1_b   = (const float*)d_in[18];
  const float* n2_w   = (const float*)d_in[19];
  const float* n2_b   = (const float*)d_in[20];
  const float* proj_w = (const float*)d_in[21];
  const float* proj_b = (const float*)d_in[22];
  float* out = (float*)d_out;

  char* ws = (char*)d_ws;
  size_t off = 0;
  auto alloc = [&](size_t bytes) -> char* {
    char* p = ws + off;
    off = (off + bytes + 255) & ~(size_t)255;
    return p;
  };
  bf16_t* nx    = (bf16_t*)alloc((size_t)ROWS * 96 * 2);
  bf16_t* v1pad = (bf16_t*)alloc((size_t)ROWS * 64 * 2);
  bf16_t* v2pad = (bf16_t*)alloc((size_t)ROWS * 64 * 2);
  float*  gate  = (float*)alloc((size_t)ROWS * 96 * 4);
  float*  xc    = (float*)alloc((size_t)ROWS * 96 * 4);
  bf16_t* valb  = (bf16_t*)alloc((size_t)ROWS * 96 * 2);
  float*  x1raw = (float*)alloc((size_t)BB * 256 * 96 * 4);
  float*  x2raw = (float*)alloc((size_t)BB * 1024 * 96 * 4);
  bf16_t* x1b   = (bf16_t*)alloc((size_t)BB * 256 * 96 * 2);
  bf16_t* x2b   = (bf16_t*)alloc((size_t)BB * 1024 * 96 * 2);
  bf16_t* q1pad = (bf16_t*)alloc((size_t)BB * 256 * 64 * 2);
  bf16_t* q2pad = (bf16_t*)alloc((size_t)BB * 1024 * 64 * 2);
  bf16_t* k1T   = (bf16_t*)alloc((size_t)BB * 48 * 256 * 2);
  bf16_t* k2T   = (bf16_t*)alloc((size_t)BB * 48 * 1024 * 2);
  bf16_t* WhT   = (bf16_t*)alloc(192 * 96 * 2);
  bf16_t* WqkT  = (bf16_t*)alloc(48 * 96 * 2);
  bf16_t* projT = (bf16_t*)alloc(96 * 96 * 2);
  bf16_t* w1r   = (bf16_t*)alloc((size_t)96 * 6144 * 2);
  bf16_t* w2r   = (bf16_t*)alloc((size_t)96 * 1536 * 2);

  auto cdiv = [](int a, int b) { return (a + b - 1) / b; };

  // weight prep
  transpose_bf16<<<cdiv(96 * 192, 256), 256, 0, stream>>>(Wh, WhT, 96, 192);
  transpose_bf16<<<cdiv(96 * 48, 256), 256, 0, stream>>>(Wqk, WqkT, 96, 48);
  transpose_bf16<<<cdiv(96 * 96, 256), 256, 0, stream>>>(proj_w, projT, 96, 96);
  conv_w_reorder<<<cdiv(96 * 6144, 256), 256, 0, stream>>>(sr1_w, w1r, 64);
  conv_w_reorder<<<cdiv(96 * 1536, 256), 256, 0, stream>>>(sr2_w, w2r, 16);

  // zero padded operand arrays (pad columns must be 0 for K-padded WMMA)
  hipMemsetAsync(v1pad, 0, (size_t)ROWS * 64 * 2, stream);
  hipMemsetAsync(v2pad, 0, (size_t)ROWS * 64 * 2, stream);
  hipMemsetAsync(q1pad, 0, (size_t)BB * 256 * 64 * 2, stream);
  hipMemsetAsync(q2pad, 0, (size_t)BB * 1024 * 64 * 2, stream);

  // 1) LayerNorm of x -> bf16 nx
  ln_kernel<<<ROWS / 8, 256, 0, stream>>>(x, norm_w, norm_b, nx, ROWS, 0);
  // 2) h = silu(nx@Wh+bh) -> v1/v2 (bf16 pad) + gate (f32)
  gemm_h<<<(ROWS / 16) * 12 / 8, 256, 0, stream>>>(nx, WhT, bh, v1pad, v2pad, gate);
  // 3) conv patch-embeds (im2col GEMM)
  conv_gemm<<<(BB * 256 / 16) * 6 / 8, 256, 0, stream>>>(nx, w1r, sr1_b, x1raw, 8, 256, 16, 6144);
  conv_gemm<<<(BB * 1024 / 16) * 6 / 8, 256, 0, stream>>>(nx, w2r, sr2_b, x2raw, 4, 1024, 32, 1536);
  // 4) LN + gelu on branch tokens
  ln_kernel<<<BB * 256 / 8, 256, 0, stream>>>(x1raw, n1_w, n1_b, x1b, BB * 256, 1);
  ln_kernel<<<BB * 1024 / 8, 256, 0, stream>>>(x2raw, n2_w, n2_b, x2b, BB * 1024, 1);
  // 5) q/k generation
  qk_kernel<<<(BB * 256 / 16) * 3 / 8, 256, 0, stream>>>(x1b, WqkT, bqk, g1, be1, q1pad, k1T, 256, BB * 256);
  qk_kernel<<<(BB * 1024 / 16) * 3 / 8, 256, 0, stream>>>(x2b, WqkT, bqk, g2, be2, q2pad, k2T, 1024, BB * 1024);
  // 6) fused attention -> xc halves
  attn_kernel<<<ROWS / 64, 128, 0, stream>>>(v1pad, q1pad, k1T, xc, 256, 0);
  attn_kernel<<<ROWS / 64, 128, 0, stream>>>(v2pad, q2pad, k2T, xc, 1024, 48);
  // 7) gating + projection + residual
  value_mul<<<cdiv(ROWS * 96, 256), 256, 0, stream>>>(xc, gate, valb, ROWS * 96);
  proj_kernel<<<(ROWS / 16) * 6 / 8, 256, 0, stream>>>(valb, projT, proj_b, xc, out);
}
